// DAGNN_16604343566803
// MI455X (gfx1250) — matrix operand order repaired and checked by reading, not compiled, not run on hardware
//
#include <hip/hip_runtime.h>
#include <hip/hip_bf16.h>
#include <math.h>

// ---------------- problem constants ----------------
#define NN    100000
#define EE    3200000
#define INC   500
#define KP    512        // padded K for GEMM1
#define HIDC  256
#define OUTC  64
#define KHOP  10

// ---------------- WMMA / TDM types ----------------
typedef __bf16        v16bf __attribute__((ext_vector_type(16)));
typedef float         v8f   __attribute__((ext_vector_type(8)));
typedef unsigned int  v4u   __attribute__((ext_vector_type(4)));
typedef int           v8i   __attribute__((ext_vector_type(8)));
typedef int           v4i   __attribute__((ext_vector_type(4)));

union Frag16 { v4u q[2]; v16bf v; };

// fp32 -> bf16 (round to nearest even-ish)
__device__ __forceinline__ unsigned short f2bf(float f) {
    unsigned int u = __float_as_uint(f);
    u += 0x7FFFu + ((u >> 16) & 1u);
    return (unsigned short)(u >> 16);
}

// Escape an LDS address so GlobalOpt cannot prove "no stores to this global":
// the TDM writes LDS through a descriptor, invisibly to the IR. Once the
// address escapes into an asm with a memory clobber, loads from it can no
// longer be folded to undef.
__device__ __forceinline__ void escape_lds(const void* p) {
    asm volatile("" :: "v"(p) : "memory");
}

// ---------------- TDM: DMA a 2D bf16 tile (global -> LDS) ----------------
// D# per CDNA5 ISA. 2-byte elements, 2D tile, no padding,
// no gather, no multicast. gptr = tile start; dims/strides in elements.
// 6-arg builtin form: (g0 v4u, g1 v8i, g2 v4i, g3 v4i, g4 v8i, cpol).
__device__ __forceinline__ void tdm_load_tile_to_lds(const void* gptr, unsigned ldsByteOff,
                                                     unsigned tile0, unsigned tile1,
                                                     unsigned tdim0, unsigned tdim1,
                                                     unsigned stride0) {
    unsigned long long ga = (unsigned long long)(uintptr_t)gptr;
    v4u g0;
    g0[0] = 1u;                                                // count=1, user, no gather
    g0[1] = ldsByteOff;                                        // lds_addr (bytes)
    g0[2] = (unsigned)ga;                                      // global_addr[31:0]
    g0[3] = (unsigned)((ga >> 32) & 0x1FFFFFFu) | (2u << 30);  // global_addr[56:32] | type=2
    v8i g1;
    g1[0] = (int)(1u << 16);                                   // data_size=1 (2B), wg_mask=0
    g1[1] = (int)(tdim0 << 16);                                // tensor_dim0[15:0]
    g1[2] = (int)((tdim0 >> 16) | (tdim1 << 16));              // tensor_dim0[31:16] | tensor_dim1[15:0]
    g1[3] = (int)((tdim1 >> 16) | (tile0 << 16));              // tensor_dim1[31:16] | tile_dim0
    g1[4] = (int)(tile1 & 0xFFFFu);                            // tile_dim1 ; tile_dim2=0
    g1[5] = (int)stride0;                                      // tensor_dim0_stride[31:0]
    g1[6] = 0;
    g1[7] = 0;
    v4i g2 = {0, 0, 0, 0};
    v4i g3 = {0, 0, 0, 0};
    v8i g4 = {0, 0, 0, 0, 0, 0, 0, 0};
    __builtin_amdgcn_tensor_load_to_lds(g0, g1, g2, g3, g4, 0);
    asm volatile("" ::: "memory");   // DMA may write LDS (escaped): keep loads alive
}

__device__ __forceinline__ void tdm_wait(int pending) {
    if (pending) __builtin_amdgcn_s_wait_tensorcnt(1);
    else         __builtin_amdgcn_s_wait_tensorcnt(0);
    asm volatile("" ::: "memory");   // loads below must not move above the wait
}

// ---------------- conversions to bf16 ----------------
__global__ void k_convertA(const float* __restrict__ nf, unsigned short* __restrict__ abf) {
    int gid = blockIdx.x * 256 + threadIdx.x;
    if (gid >= NN * KP) return;
    int n = gid >> 9, k = gid & (KP - 1);
    abf[gid] = (k < INC) ? f2bf(nf[(size_t)n * INC + k]) : (unsigned short)0;
}
__global__ void k_convertW1(const float* __restrict__ W1, unsigned short* __restrict__ w1bf) {
    int gid = blockIdx.x * 256 + threadIdx.x;
    if (gid >= HIDC * KP) return;
    int n = gid >> 9, k = gid & (KP - 1);
    w1bf[gid] = (k < INC) ? f2bf(W1[(size_t)n * INC + k]) : (unsigned short)0;
}
__global__ void k_convertW2(const float* __restrict__ W2, unsigned short* __restrict__ w2bf) {
    int gid = blockIdx.x * 256 + threadIdx.x;
    if (gid >= OUTC * HIDC) return;
    w2bf[gid] = f2bf(W2[gid]);
}

// ---------------- GEMM1: hid = relu(A @ W1^T + b1), bf16 WMMA ----------------
// 8 waves/block; wave -> 16 rows x 256 cols. W1 bf16 streamed into a 2x64KB LDS
// double buffer by the Tensor Data Mover while waves compute on the other buffer.
__global__ __launch_bounds__(256) void k_gemm1(const unsigned short* __restrict__ abf,
                                               const unsigned short* __restrict__ w1bf,
                                               const float* __restrict__ b1,
                                               unsigned short* __restrict__ hid) {
    __shared__ unsigned short sW[2][HIDC * 128];   // 2 x 64 KB
    escape_lds(&sW[0][0]);                         // DMA target aliases this
    const int tid  = threadIdx.x;
    const int wave = tid >> 5;
    const int lane = tid & 31;
    const int ln   = lane & 15;
    const int lh   = lane >> 4;
    const int m0   = (blockIdx.x * 8 + wave) * 16;

    // prologue: DMA chunk 0 into buffer 0 (wave 0 only; TDM is wave-scalar)
    if (wave == 0)
        tdm_load_tile_to_lds(w1bf, 0u, 128u, HIDC, KP, HIDC, KP);

    v8f c[16] = {};
    int arow = m0 + ln; if (arow > NN - 1) arow = NN - 1;
    const unsigned short* aptr = abf + (size_t)arow * KP;

    for (int ch = 0; ch < 4; ++ch) {
        if (wave == 0) {
            if (ch + 1 < 4) {  // prefetch next chunk into the other buffer
                tdm_load_tile_to_lds(w1bf + (ch + 1) * 128,
                                     ((ch + 1) & 1) ? 65536u : 0u,
                                     128u, HIDC, KP, HIDC, KP);
                // TDM completes in-order: <=1 outstanding => chunk `ch` landed
                tdm_wait(1);
            } else {
                tdm_wait(0);
            }
        }
        __syncthreads();   // publish buffer ch&1 to all waves
        const unsigned short* sw = sW[ch & 1];

        #pragma unroll
        for (int k0 = 0; k0 < 128; k0 += 32) {
            Frag16 a;
            const int kg = ch * 128 + k0 + lh * 16;
            a.q[0] = *(const v4u*)(aptr + kg);
            a.q[1] = *(const v4u*)(aptr + kg + 8);
            Frag16 bcur;
            {
                const unsigned short* bp = sw + ln * 128 + k0 + lh * 16;
                bcur.q[0] = *(const v4u*)bp;
                bcur.q[1] = *(const v4u*)(bp + 8);
            }
            #pragma unroll
            for (int t = 0; t < 16; ++t) {
                Frag16 bnext = {};
                if (t < 15) {   // software-pipeline the DS loads past the WMMA
                    const unsigned short* bp = sw + ((t + 1) * 16 + ln) * 128 + k0 + lh * 16;
                    bnext.q[0] = *(const v4u*)bp;
                    bnext.q[1] = *(const v4u*)(bp + 8);
                }
                c[t] = __builtin_amdgcn_wmma_f32_16x16x32_bf16(
                        false, a.v, false, bcur.v, (short)0, c[t], false, false);
                bcur = bnext;
            }
        }
        __syncthreads();   // reads done before this buffer is DMA-overwritten
    }

    #pragma unroll
    for (int t = 0; t < 16; ++t) {
        int n = t * 16 + ln;
        float bias = b1[n];
        #pragma unroll
        for (int r = 0; r < 8; ++r) {
            int m = m0 + lh * 8 + r;
            if (m < NN) {
                float v = c[t][r] + bias;
                hid[(size_t)m * HIDC + n] = f2bf(v > 0.0f ? v : 0.0f);
            }
        }
    }
}

// ---------------- GEMM2: x0 = hid @ W2^T + b2 (fp32 out) ----------------
__global__ __launch_bounds__(256) void k_gemm2(const unsigned short* __restrict__ hid,
                                               const unsigned short* __restrict__ w2bf,
                                               const float* __restrict__ b2,
                                               float* __restrict__ x0) {
    __shared__ unsigned short sW[OUTC * HIDC];  // 32 KB, whole W2
    escape_lds(&sW[0]);                         // DMA target aliases this
    const int tid  = threadIdx.x;
    const int wave = tid >> 5;
    const int lane = tid & 31;
    const int ln   = lane & 15;
    const int lh   = lane >> 4;
    const int m0   = (blockIdx.x * 8 + wave) * 16;

    if (wave == 0) {
        tdm_load_tile_to_lds(w2bf, 0u, HIDC, OUTC, HIDC, OUTC, HIDC);
        tdm_wait(0);
    }
    __syncthreads();

    v8f c[4] = {};
    int arow = m0 + ln; if (arow > NN - 1) arow = NN - 1;
    const unsigned short* aptr = hid + (size_t)arow * HIDC;

    #pragma unroll
    for (int k0 = 0; k0 < HIDC; k0 += 32) {
        Frag16 a;
        const int kg = k0 + lh * 16;
        a.q[0] = *(const v4u*)(aptr + kg);
        a.q[1] = *(const v4u*)(aptr + kg + 8);
        Frag16 bcur;
        {
            const unsigned short* bp = sW + ln * HIDC + k0 + lh * 16;
            bcur.q[0] = *(const v4u*)bp;
            bcur.q[1] = *(const v4u*)(bp + 8);
        }
        #pragma unroll
        for (int t = 0; t < 4; ++t) {
            Frag16 bnext = {};
            if (t < 3) {
                const unsigned short* bp = sW + ((t + 1) * 16 + ln) * HIDC + k0 + lh * 16;
                bnext.q[0] = *(const v4u*)bp;
                bnext.q[1] = *(const v4u*)(bp + 8);
            }
            c[t] = __builtin_amdgcn_wmma_f32_16x16x32_bf16(
                    false, a.v, false, bcur.v, (short)0, c[t], false, false);
            bcur = bnext;
        }
    }

    #pragma unroll
    for (int t = 0; t < 4; ++t) {
        int n = t * 16 + ln;
        float bias = b2[n];
        #pragma unroll
        for (int r = 0; r < 8; ++r) {
            int m = m0 + lh * 8 + r;
            if (m < NN) x0[(size_t)m * OUTC + n] = c[t][r] + bias;
        }
    }
}

// ---------------- gcn_norm pieces ----------------
__global__ void k_zero_deg(int* __restrict__ deg) {
    int gid = blockIdx.x * 256 + threadIdx.x;
    if (gid < NN) deg[gid] = 0;
}
__global__ void k_deg(const int* __restrict__ ei, int* __restrict__ deg) {
    int e = blockIdx.x * 256 + threadIdx.x;
    if (e >= EE) return;
    int r = ei[e], c = ei[EE + e];
    if (r != c) atomicAdd(&deg[c], 1);
}
__global__ void k_dinv(const int* __restrict__ deg, float* __restrict__ dinv) {
    int gid = blockIdx.x * 256 + threadIdx.x;
    if (gid < NN) dinv[gid] = rsqrtf((float)deg[gid] + 1.0f);
}
__global__ void k_norm(const int* __restrict__ ei, const float* __restrict__ dinv,
                       float* __restrict__ nrm) {
    int e = blockIdx.x * 256 + threadIdx.x;
    if (e >= EE) return;
    int r = ei[e], c = ei[EE + e];
    nrm[e] = (r != c) ? dinv[r] * dinv[c] : 0.0f;
}

// ---------------- propagation ----------------
__global__ void k_self(const float* __restrict__ dinv, const float* __restrict__ x,
                       float* __restrict__ y) {
    int gid = blockIdx.x * 256 + threadIdx.x;
    if (gid >= NN * OUTC) return;
    float d = dinv[gid >> 6];
    y[gid] = d * d * x[gid];
}
__global__ void k_edges(const int* __restrict__ ei, const float* __restrict__ nrm,
                        const float* __restrict__ x, float* __restrict__ y) {
    int gid = blockIdx.x * 256 + threadIdx.x;   // EE*32 threads total
    int e = gid >> 5;
    if (e >= EE) return;
    float nv = nrm[e];
    if (nv == 0.0f) return;
    int r = ei[e], c = ei[EE + e];
    int ch = (gid & 31) * 2;
    const float* xp = x + (size_t)r * OUTC + ch;
    float v0 = xp[0], v1 = xp[1];
    float* yp = y + (size_t)c * OUTC + ch;
    atomicAdd(yp,     nv * v0);
    atomicAdd(yp + 1, nv * v1);
}

// ---------------- fused sigmoid-attention accumulate ----------------
__global__ void k_attn(const float* __restrict__ x, const float* __restrict__ pw,
                       const float* __restrict__ pb, float* __restrict__ out, float beta) {
    int gid = blockIdx.x * 256 + threadIdx.x;   // NN*32 threads
    int node = gid >> 5;
    if (node >= NN) return;
    int l = gid & 31;
    const float* xp = x + (size_t)node * OUTC;
    float x0 = xp[l], x1 = xp[l + 32];
    float s = x0 * pw[l] + x1 * pw[l + 32];
    #pragma unroll
    for (int m = 16; m >= 1; m >>= 1) s += __shfl_xor(s, m, 32);
    float score = 1.0f / (1.0f + __expf(-(s + pb[0])));
    float* op = out + (size_t)node * OUTC;
    op[l]      = beta * op[l]      + score * x0;
    op[l + 32] = beta * op[l + 32] + score * x1;
}

// ---------------- host launcher ----------------
extern "C" void kernel_launch(void* const* d_in, const int* in_sizes, int n_in,
                              void* d_out, int out_size, void* d_ws, size_t ws_size,
                              hipStream_t stream) {
    (void)in_sizes; (void)n_in; (void)out_size; (void)ws_size;
    const float* nf = (const float*)d_in[0];
    const int*   ei = (const int*)  d_in[1];
    const float* W1 = (const float*)d_in[2];
    const float* b1 = (const float*)d_in[3];
    const float* W2 = (const float*)d_in[4];
    const float* b2 = (const float*)d_in[5];
    const float* pw = (const float*)d_in[6];
    const float* pb = (const float*)d_in[7];
    float* out = (float*)d_out;

    char* w = (char*)d_ws;
    size_t off = 0;
    auto carve = [&](size_t bytes) {
        void* p = w + off;
        off += (bytes + 255) & ~(size_t)255;
        return p;
    };
    unsigned short* abf  = (unsigned short*)carve((size_t)NN * KP * 2);     // 102.4 MB
    unsigned short* hid  = (unsigned short*)carve((size_t)NN * HIDC * 2);   //  51.2 MB
    unsigned short* w1bf = (unsigned short*)carve((size_t)HIDC * KP * 2);   //  0.25 MB
    unsigned short* w2bf = (unsigned short*)carve((size_t)OUTC * HIDC * 2); //  32 KB
    float*          xA   = (float*)carve((size_t)NN * OUTC * 4);            //  25.6 MB
    float*          xB   = (float*)carve((size_t)NN * OUTC * 4);            //  25.6 MB
    int*            deg  = (int*)  carve((size_t)NN * 4);
    float*          dinv = (float*)carve((size_t)NN * 4);
    float*          nrm  = (float*)carve((size_t)EE * 4);                   //  12.8 MB

    const int T = 256;
    // bf16 conversions
    k_convertA <<<(NN * KP + T - 1) / T, T, 0, stream>>>(nf, abf);
    k_convertW1<<<(HIDC * KP + T - 1) / T, T, 0, stream>>>(W1, w1bf);
    k_convertW2<<<(OUTC * HIDC + T - 1) / T, T, 0, stream>>>(W2, w2bf);
    // MLP (WMMA bf16, TDM-staged weights)
    k_gemm1<<<(NN + 127) / 128, T, 0, stream>>>(abf, w1bf, b1, hid);
    k_gemm2<<<(NN + 127) / 128, T, 0, stream>>>(hid, w2bf, b2, xA);

    // gcn_norm
    k_zero_deg<<<(NN + T - 1) / T, T, 0, stream>>>(deg);
    k_deg<<<(EE + T - 1) / T, T, 0, stream>>>(ei, deg);
    k_dinv<<<(NN + T - 1) / T, T, 0, stream>>>(deg, dinv);
    k_norm<<<(EE + T - 1) / T, T, 0, stream>>>(ei, dinv, nrm);

    // hop 0 contribution (write, beta=0)
    k_attn<<<(NN * 32 + T - 1) / T, T, 0, stream>>>(xA, pw, pb, out, 0.0f);

    // K hops, fused attention accumulate, ping-pong
    float* xs = xA; float* xd = xB;
    for (int k = 0; k < KHOP; ++k) {
        k_self <<<(NN * OUTC + T - 1) / T, T, 0, stream>>>(dinv, xs, xd);
        k_edges<<<(EE * 32 / T), T, 0, stream>>>(ei, nrm, xs, xd);
        k_attn <<<(NN * 32 + T - 1) / T, T, 0, stream>>>(xd, pw, pb, out, 1.0f);
        float* tmp = xs; xs = xd; xd = tmp;
    }
}
